// GatedGraph_35802847379877
// MI455X (gfx1250) — matrix operand order repaired and checked by reading, not compiled, not run on hardware
//
#include <hip/hip_runtime.h>
#include <math.h>

typedef __bf16 bf16;
typedef __attribute__((ext_vector_type(16))) __bf16 v16bf;
typedef __attribute__((ext_vector_type(8)))  __bf16 v8bf;
typedef __attribute__((ext_vector_type(8)))  float  v8f;

#define NODES 4000
#define SZ 64
#define TT 12

// ---- fragment builders -----------------------------------------------------
// A (16x32 bf16, M=lane&15): elems 0..7 -> K = k0 + 8*(lane>>4) + e
//                            elems 8..15 -> K = k0 + 16 + 8*(lane>>4) + e
// B (32x16 bf16, N=lane&15): elems 0..7 -> K = k0 + 8*(lane>>4) + e
//                            elems 8..15 -> K = k0 + 16 + 8*(lane>>4) + e
__device__ inline v16bf combine(v8bf lo, v8bf hi) {
    v16bf r;
#pragma unroll
    for (int i = 0; i < 8; ++i) { r[i] = lo[i]; r[i + 8] = hi[i]; }
    return r;
}
__device__ inline v16bf frag_a(const bf16* row, int k0, int lhi) {
    const bf16* p = row + k0 + 8 * lhi;
    return combine(*(const v8bf*)p, *(const v8bf*)(p + 16));
}
__device__ inline v16bf frag_a_nt(const bf16* row, int k0, int lhi) {
    const bf16* p = row + k0 + 8 * lhi;
    v8bf lo = __builtin_nontemporal_load((const v8bf*)p);
    v8bf hi = __builtin_nontemporal_load((const v8bf*)(p + 16));
    return combine(lo, hi);
}
__device__ inline v16bf frag_b(const bf16* row, int k0, int lhi) {
    const bf16* p = row + k0 + 8 * lhi;
    return combine(*(const v8bf*)p, *(const v8bf*)(p + 16));
}
#define WMMA_BF16(a, b, c) \
    __builtin_amdgcn_wmma_f32_16x16x32_bf16(false, (a), false, (b), (short)0, (c), false, false)

// ---- one-time preprocessing ------------------------------------------------
// Transpose+convert edge_matrix: E[t][n][m] fp32 -> ET[t][m][n] bf16 (384 MB).
__global__ __launch_bounds__(256) void prep_E(const float* __restrict__ E,
                                              bf16* __restrict__ ET) {
    __shared__ bf16 tile[32][33];
    const int t  = blockIdx.z;
    const int nb = blockIdx.x * 32, mb = blockIdx.y * 32;
    const int lx = threadIdx.x, ly = threadIdx.y;
    const float* src = E + (size_t)t * NODES * NODES;
    bf16* dst = ET + (size_t)t * NODES * NODES;
#pragma unroll
    for (int r = 0; r < 32; r += 8)
        tile[ly + r][lx] = (bf16)src[(size_t)(nb + ly + r) * NODES + mb + lx];
    __syncthreads();
#pragma unroll
    for (int r = 0; r < 32; r += 8)
        dst[(size_t)(mb + ly + r) * NODES + nb + lx] = tile[lx][ly + r];
}

// Transpose+convert the small weights: W[t][k][n]->WT[t][n][k], uz_ur, uh.
__global__ void prep_w(const float* __restrict__ W, const float* __restrict__ uzur,
                       const float* __restrict__ uh, bf16* __restrict__ WT,
                       bf16* __restrict__ uzurT, bf16* __restrict__ uhT) {
    int i = blockIdx.x * blockDim.x + threadIdx.x;
    const int nW = TT * SZ * 192, nU = SZ * 128, nH = SZ * SZ;
    if (i < nW) {
        int t = i / (SZ * 192), r = i % (SZ * 192), k = r / 192, n = r % 192;
        WT[((size_t)t * 192 + n) * SZ + k] = (bf16)W[i];
    } else if (i < nW + nU) {
        int j = i - nW, k = j / 128, n = j % 128;
        uzurT[n * SZ + k] = (bf16)uzur[j];
    } else if (i < nW + nU + nH) {
        int j = i - nW - nU, k = j / SZ, n = j % SZ;
        uhT[n * SZ + k] = (bf16)uh[j];
    }
}

// ---- per-iteration kernels -------------------------------------------------
// h fp32 -> h_bf16[m][s] (A operand) and hT_bf16[s][m] (B operand, L2-resident)
__global__ void convert_h(const float* __restrict__ hin, bf16* __restrict__ hbf,
                          bf16* __restrict__ hT) {
    int i = blockIdx.x * blockDim.x + threadIdx.x;
    if (i < NODES * SZ) {
        int m = i >> 6, s = i & 63;
        float v = hin[i];
        hbf[i] = (bf16)v;
        hT[s * NODES + m] = (bf16)v;
    }
}

// act[t] = E_t^T @ h + ba[t]   (M=4000, N=64, K=4000) -> bf16
// One wave per (64-row M-block, edge type): 4 A tiles share 4 B fragments,
// so the L2-resident hT stream is read 4x less. ET is streamed with TH=NT.
__global__ __launch_bounds__(32) void gemm_act(const bf16* __restrict__ ET,
                                               const bf16* __restrict__ hT,
                                               const float* __restrict__ ba,
                                               bf16* __restrict__ act) {
    const int lane = threadIdx.x, l15 = lane & 15, lhi = lane >> 4;
    const int m0 = blockIdx.x * 64, t = blockIdx.y;

    const bf16* Arow[4];
#pragma unroll
    for (int i = 0; i < 4; ++i) {
        int m = m0 + 16 * i + l15;
        if (m >= NODES) m = NODES - 1;  // clamp (stores are guarded)
        Arow[i] = ET + ((size_t)t * NODES + m) * NODES;
    }

    v8f acc[4][4] = {};
    for (int k = 0; k < NODES; k += 32) {
        v16bf b[4];
#pragma unroll
        for (int j = 0; j < 4; ++j)
            b[j] = frag_b(hT + (size_t)(16 * j + l15) * NODES, k, lhi);
#pragma unroll
        for (int i = 0; i < 4; ++i) {
            v16bf a = frag_a_nt(Arow[i], k, lhi);
#pragma unroll
            for (int j = 0; j < 4; ++j) acc[i][j] = WMMA_BF16(a, b[j], acc[i][j]);
        }
    }
#pragma unroll
    for (int i = 0; i < 4; ++i)
#pragma unroll
        for (int j = 0; j < 4; ++j) {
            float bav = ba[t * SZ + 16 * j + l15];
#pragma unroll
            for (int v = 0; v < 8; ++v) {
                int m = m0 + 16 * i + v + 8 * lhi;
                if (m < NODES)
                    act[((size_t)t * NODES + m) * SZ + 16 * j + l15] =
                        (bf16)(acc[i][j][v] + bav);
            }
        }
}

// Fused: awzrh = sum_t act_t@W_t + 12*bw, += h@Uzr (z|r cols), GRU gates,
// hh = tanh(awh + (r*h)@Uh), h' = (1-z)h + z*hh.  One wave per 16 rows.
__global__ __launch_bounds__(32) void proj_gru(const bf16* __restrict__ act,
                                               const bf16* __restrict__ WT,
                                               const bf16* __restrict__ uzurT,
                                               const bf16* __restrict__ uhT,
                                               const bf16* __restrict__ hbf,
                                               const float* __restrict__ bw,
                                               const float* __restrict__ hin,
                                               float* __restrict__ hout) {
    __shared__ bf16 rh[16 * SZ];
    const int lane = threadIdx.x, l15 = lane & 15, lhi = lane >> 4;
    const int m0 = blockIdx.x * 16;

    v8f acc[12];
#pragma unroll
    for (int j = 0; j < 12; ++j) {
        float b = 12.0f * bw[16 * j + l15];  // bw broadcast over T, then summed
#pragma unroll
        for (int v = 0; v < 8; ++v) acc[j][v] = b;
    }
    // sum over edge types: act_t(16x64) @ W_t(64x192)
    for (int t = 0; t < TT; ++t) {
        const bf16* Arow = act + ((size_t)t * NODES + m0 + l15) * SZ;
#pragma unroll
        for (int c = 0; c < 2; ++c) {
            v16bf a = frag_a(Arow, 32 * c, lhi);
#pragma unroll
            for (int j = 0; j < 12; ++j) {
                v16bf b = frag_b(WT + ((size_t)t * 192 + 16 * j + l15) * SZ, 32 * c, lhi);
                acc[j] = WMMA_BF16(a, b, acc[j]);
            }
        }
    }
    // h(16x64) @ Uzr(64x128) accumulated straight into z/r columns
    {
        const bf16* Arow = hbf + (size_t)(m0 + l15) * SZ;
#pragma unroll
        for (int c = 0; c < 2; ++c) {
            v16bf a = frag_a(Arow, 32 * c, lhi);
#pragma unroll
            for (int j = 0; j < 8; ++j) {
                v16bf b = frag_b(uzurT + (size_t)(16 * j + l15) * SZ, 32 * c, lhi);
                acc[j] = WMMA_BF16(a, b, acc[j]);
            }
        }
    }
    // gates on the C-fragment layout: row m = m0+v+8*lhi, col s = 16j+l15
    float z[4][8], hv[4][8];
#pragma unroll
    for (int j = 0; j < 4; ++j)
#pragma unroll
        for (int v = 0; v < 8; ++v) {
            int m = m0 + v + 8 * lhi, s = 16 * j + l15;
            float h = hin[m * SZ + s];
            hv[j][v] = h;
            z[j][v] = 1.0f / (1.0f + __expf(-acc[j][v]));
            float r = 1.0f / (1.0f + __expf(-acc[j + 4][v]));
            rh[(v + 8 * lhi) * SZ + s] = (bf16)(r * h);
        }
    __syncthreads();
    // hh = tanh(awh + (r*h) @ Uh): re-layout via LDS, accumulate into awh frags
#pragma unroll
    for (int c = 0; c < 2; ++c) {
        v16bf a = frag_a(&rh[l15 * SZ], 32 * c, lhi);
#pragma unroll
        for (int j = 0; j < 4; ++j) {
            v16bf b = frag_b(uhT + (size_t)(16 * j + l15) * SZ, 32 * c, lhi);
            acc[8 + j] = WMMA_BF16(a, b, acc[8 + j]);
        }
    }
#pragma unroll
    for (int j = 0; j < 4; ++j)
#pragma unroll
        for (int v = 0; v < 8; ++v) {
            int m = m0 + v + 8 * lhi, s = 16 * j + l15;
            float hh = tanhf(acc[8 + j][v]);
            hout[m * SZ + s] = (1.0f - z[j][v]) * hv[j][v] + z[j][v] * hh;
        }
}

// ---- host ------------------------------------------------------------------
extern "C" void kernel_launch(void* const* d_in, const int* in_sizes, int n_in,
                              void* d_out, int out_size, void* d_ws, size_t ws_size,
                              hipStream_t stream) {
    const float* x    = (const float*)d_in[0];
    const float* E    = (const float*)d_in[1];
    const float* ba   = (const float*)d_in[2];
    const float* bw   = (const float*)d_in[3];
    const float* W    = (const float*)d_in[4];
    const float* uzur = (const float*)d_in[5];
    const float* uh   = (const float*)d_in[6];
    float* out = (float*)d_out;

    char* p = (char*)d_ws;
    auto alloc = [&](size_t bytes) -> char* {
        char* r = p;
        p += (bytes + 255) & ~(size_t)255;
        return r;
    };
    bf16* ET    = (bf16*)alloc((size_t)TT * NODES * NODES * 2);  // 384 MB
    bf16* WT    = (bf16*)alloc((size_t)TT * 192 * SZ * 2);
    bf16* uzurT = (bf16*)alloc(128 * SZ * 2);
    bf16* uhT   = (bf16*)alloc(SZ * SZ * 2);
    bf16* hbf   = (bf16*)alloc((size_t)NODES * SZ * 2);
    bf16* hT    = (bf16*)alloc((size_t)SZ * NODES * 2);
    bf16* actb  = (bf16*)alloc((size_t)TT * NODES * SZ * 2);
    float* hb0  = (float*)alloc((size_t)NODES * SZ * 4);
    float* hb1  = (float*)alloc((size_t)NODES * SZ * 4);

    const int nSmall = TT * SZ * 192 + SZ * 128 + SZ * SZ;
    prep_w<<<(nSmall + 255) / 256, 256, 0, stream>>>(W, uzur, uh, WT, uzurT, uhT);
    prep_E<<<dim3(125, 125, TT), dim3(32, 8), 0, stream>>>(E, ET);

    const float* hcur = x;
    for (int it = 0; it < 10; ++it) {
        convert_h<<<(NODES * SZ + 255) / 256, 256, 0, stream>>>(hcur, hbf, hT);
        gemm_act<<<dim3((NODES + 63) / 64, TT), 32, 0, stream>>>(ET, hT, ba, actb);
        float* hnext = (it == 9) ? out : ((it & 1) ? hb1 : hb0);
        proj_gru<<<NODES / 16, 32, 0, stream>>>(actb, WT, uzurT, uhT, hbf, bw, hcur, hnext);
        hcur = hnext;
    }
}
